// SparseMiddleEncoder_36026185679343
// MI455X (gfx1250) — matrix operand order repaired and checked by reading, not compiled, not run on hardware
//
#include <hip/hip_runtime.h>
#include <hip/hip_bf16.h>
#include <stdint.h>
#include <stddef.h>

// ---------------------------------------------------------------------------
// SparseMiddleEncoder for MI455X (gfx1250).
// Implicit-GEMM over the 27 conv taps with v_wmma_f32_16x16x32_bf16.
// Channels-last bf16 staging WITH a zero halo in H/W: the inner loop is pure
// {global_load_b128 x4 -> wmma} with full EXEC, all offsets compile-time.
// 192MB L2 keeps the staged tensor resident across the 27-tap reuse.
// ---------------------------------------------------------------------------

typedef __attribute__((ext_vector_type(16))) __bf16 v16bf;
typedef __attribute__((ext_vector_type(8)))  __bf16 v8bf;
typedef __attribute__((ext_vector_type(8)))  float  v8f;
typedef __attribute__((ext_vector_type(4)))  float  v4f;
typedef __attribute__((ext_vector_type(4)))  uint   v4u;

union FragBF { v16bf v; v8bf h[2]; };

constexpr int H  = 200;
constexpr int W  = 176;
constexpr int HW = H * W;
constexpr int HP = H + 2;      // padded (halo) height
constexpr int WP = W + 2;      // padded (halo) width
constexpr int HWP = HP * WP;
constexpr int COUT = 64;

// ---------------- generic zero fill (16B granules) ------------------------
__global__ void zero_b128(v4u* __restrict__ p, int n16)
{
    int i = blockIdx.x * blockDim.x + threadIdx.x;
    if (i < n16) p[i] = v4u{0u, 0u, 0u, 0u};
}

// ---------------- prep: BN constant folding -------------------------------
__global__ void prep_bn(const float* __restrict__ b,  const float* __restrict__ g,
                        const float* __restrict__ be, const float* __restrict__ rm,
                        const float* __restrict__ rv,
                        float* __restrict__ alpha, float* __restrict__ beta)
{
    int c = threadIdx.x;
    if (c < COUT) {
        float a = g[c] * rsqrtf(rv[c] + 1e-5f);
        alpha[c] = a;
        beta[c]  = (b[c] - rm[c]) * a + be[c];
    }
}

// ---------------- prep: weights [co][ci][27] f32 -> [t][co][ci] bf16 ------
__global__ void prep_w(const float* __restrict__ w, __bf16* __restrict__ wt, int Cin)
{
    int idx = blockIdx.x * blockDim.x + threadIdx.x;
    int total = 27 * COUT * Cin;
    if (idx >= total) return;
    int t  = idx / (COUT * Cin);
    int r  = idx % (COUT * Cin);
    int co = r / Cin;
    int ci = r % Cin;
    wt[idx] = (__bf16)w[((size_t)co * Cin + ci) * 27 + t];
}

// ------- prep: x (masked) NCDHW f32 -> padded NDHWC bf16 (zero halo) ------
__global__ void prep_x_pad(const float* __restrict__ x,
                           const unsigned char* __restrict__ mask,
                           __bf16* __restrict__ xP, int D)
{
    int s = blockIdx.x * blockDim.x + threadIdx.x;          // padded spatial
    if (s >= D * HWP) return;
    int wp = s % WP;
    int hp = (s / WP) % HP;
    int d  = s / HWP;
    __bf16* dst = xP + (size_t)s * 128;
    bool interior = (wp >= 1) && (wp <= W) && (hp >= 1) && (hp <= H);
    int s0 = interior ? ((d * H + (hp - 1)) * W + (wp - 1)) : 0;
    if (interior && mask[s0]) {          // ~5% occupancy: skip reads if empty
        int DHW = D * HW;
        for (int c = 0; c < 128; c += 8) {
            v8bf pk;
#pragma unroll
            for (int j = 0; j < 8; ++j)
                pk[j] = (__bf16)x[(size_t)(c + j) * DHW + s0];
            *(v8bf*)(dst + c) = pk;
        }
    } else {
        v8bf z = {};
        for (int c = 0; c < 128; c += 8) *(v8bf*)(dst + c) = z;
    }
}

// ---------------- occupancy: bool mask -> float (unpadded) ----------------
__global__ void mask_to_occ(const unsigned char* __restrict__ m, float* __restrict__ occ, int n)
{
    int i = blockIdx.x * blockDim.x + threadIdx.x;
    if (i < n) occ[i] = m[i] ? 1.0f : 0.0f;
}

// ---------------- occupancy propagation (ones-kernel conv > 0) ------------
__global__ void occ_prop(const float* __restrict__ in, float* __restrict__ out,
                         int Din, int Dout, int sd, int pd)
{
    int idx = blockIdx.x * blockDim.x + threadIdx.x;
    if (idx >= Dout * HW) return;
    int ow = idx % W;
    int oh = (idx / W) % H;
    int od = idx / HW;
    float r = 0.0f;
    for (int kd = 0; kd < 3; ++kd) {
        int id = od * sd - pd + kd;
        if ((unsigned)id >= (unsigned)Din) continue;
        for (int kh = 0; kh < 3; ++kh) {
            int ih = oh - 1 + kh;
            if ((unsigned)ih >= (unsigned)H) continue;
            for (int kw = 0; kw < 3; ++kw) {
                int iw = ow - 1 + kw;
                if ((unsigned)iw >= (unsigned)W) continue;
                if (in[(size_t)(id * H + ih) * W + iw] != 0.0f) r = 1.0f;
            }
        }
    }
    out[idx] = r;
}

// ---------------- conv3x3x3 via WMMA implicit GEMM ------------------------
// Block = 256 thr = 8 waves = 4 M-tiles (64 co) x 2 N-tiles (32 ow).
// All layer geometry is compile-time: addresses fold to immediate offsets.
// xP : [DIN][HP][WP][CIN] bf16 with zero halo in H/W.
// wt : [27][64][CIN] bf16.
// FINAL=false -> [DOUT][HP][WP][64] bf16 (halo pre-zeroed by caller);
// FINAL=true  -> f32 NCDHW straight to d_out.
template <int NCHUNK, int DIN, int DOUT, int SD, int PD, bool FINAL>
__global__ __launch_bounds__(256)
void conv3x3x3_wmma(const __bf16* __restrict__ xP,
                    const __bf16* __restrict__ wt,
                    const float*  __restrict__ alpha,
                    const float*  __restrict__ beta,
                    const float*  __restrict__ occ,   // [DOUT][H][W] unpadded
                    void* __restrict__ outp)
{
    constexpr int CIN = NCHUNK * 32;

    const int tid   = threadIdx.x;
    const int lane  = tid & 31;
    const int wave  = tid >> 5;
    const int mtile = wave & 3;        // 0..3 -> co block of 16
    const int ntile = wave >> 2;       // 0..1 -> ow block of 16
    const int l15   = lane & 15;
    const int khalf = lane >> 4;       // which K half-group this lane holds

    const int od = blockIdx.z;
    const int oh = blockIdx.y;
    const int ow = blockIdx.x * 32 + ntile * 16 + l15;
    const int owc = (ow < W) ? ow : (W - 1);   // clamp for loads; stores guarded

    // A fragment: lane = row M; 16 contiguous K per lane, base = khalf*16
    const __bf16* aBase = wt + ((size_t)(mtile * 16 + l15)) * CIN + khalf * 16;
    // B fragment: lane = col N; 16 contiguous K per lane, base = khalf*16
    // Padded coords: ih_p = oh + kh, iw_p = owc + kw  (always in range).
    const __bf16* bBase = xP + ((size_t)oh * WP + owc) * CIN + khalf * 16;

    v8f acc = {};

#pragma unroll
    for (int kd = 0; kd < 3; ++kd) {
        const int id = od * SD - PD + kd;
        if ((unsigned)id >= (unsigned)DIN) continue;   // uniform, 3x per block
        const __bf16* plane = bBase + (size_t)id * (HWP * CIN);
#pragma unroll
        for (int kh = 0; kh < 3; ++kh) {
#pragma unroll
            for (int kw = 0; kw < 3; ++kw) {
                const __bf16* bp = plane + (kh * WP + kw) * CIN;
                const __bf16* ap = aBase + (size_t)((kd * 3 + kh) * 3 + kw) * (COUT * CIN);
#pragma unroll
                for (int cc = 0; cc < NCHUNK; ++cc) {
                    FragBF A, B;
                    A.h[0] = *(const v8bf*)(ap + cc * 32);
                    A.h[1] = *(const v8bf*)(ap + cc * 32 + 8);
                    B.h[0] = *(const v8bf*)(bp + cc * 32);
                    B.h[1] = *(const v8bf*)(bp + cc * 32 + 8);
                    acc = __builtin_amdgcn_wmma_f32_16x16x32_bf16(
                        false, A.v, false, B.v, (short)0, acc, false, false);
                }
            }
        }
    }

    // Epilogue: VGPR v holds M = v + khalf*8 (within mtile), N = l15.
    if (ow < W) {
        const int cobase = mtile * 16 + khalf * 8;
        const float oc = occ[(size_t)(od * H + oh) * W + ow];
        const v4f a0 = *(const v4f*)(alpha + cobase);
        const v4f a1 = *(const v4f*)(alpha + cobase + 4);
        const v4f b0 = *(const v4f*)(beta + cobase);
        const v4f b1 = *(const v4f*)(beta + cobase + 4);
        float av[8] = { a0.x, a0.y, a0.z, a0.w, a1.x, a1.y, a1.z, a1.w };
        float bv[8] = { b0.x, b0.y, b0.z, b0.w, b1.x, b1.y, b1.z, b1.w };
        if (FINAL) {
            float* out = (float*)outp;
            const size_t sp = (size_t)od * HW + (size_t)oh * W + ow;
#pragma unroll
            for (int v = 0; v < 8; ++v) {
                float val = fmaxf(acc[v] * av[v] + bv[v], 0.0f) * oc;
                out[(size_t)(cobase + v) * (DOUT * HW) + sp] = val;
            }
        } else {
            __bf16* out = (__bf16*)outp;
            const size_t spp = ((size_t)od * HP + (oh + 1)) * WP + (ow + 1);
            v8bf pk;
#pragma unroll
            for (int v = 0; v < 8; ++v) {
                float val = fmaxf(acc[v] * av[v] + bv[v], 0.0f) * oc;
                pk[v] = (__bf16)val;
            }
            *(v8bf*)(out + spp * COUT + cobase) = pk;   // padded channels-last
        }
    }
}

// ---------------- host-side orchestration ---------------------------------
// workspace layout (bytes, 256B aligned); total ~132.6 MB
static constexpr size_t SZ_XP   = (size_t)10 * HWP * 128 * 2;  // 92,047,360
static constexpr size_t SZ_Y0P  = (size_t)5  * HWP * 64  * 2;  // 23,011,840
static constexpr size_t SZ_Y1P  = (size_t)3  * HWP * 64  * 2;  // 13,807,104
static constexpr size_t OFF_XP   = 0;
static constexpr size_t OFF_Y0P  = OFF_XP  + SZ_XP;
static constexpr size_t OFF_Y1P  = OFF_Y0P + SZ_Y0P;
static constexpr size_t OFF_W0T  = OFF_Y1P + SZ_Y1P;           // 442,368
static constexpr size_t OFF_W1T  = OFF_W0T + 27 * 64 * 128 * 2; // 221,184
static constexpr size_t OFF_W2T  = OFF_W1T + 27 * 64 * 64 * 2;
static constexpr size_t OFF_OCC0 = OFF_W2T + 27 * 64 * 64 * 2; // 10*HW*4
static constexpr size_t OFF_OCCA = OFF_OCC0 + (size_t)10 * HW * 4;
static constexpr size_t OFF_OCCB = OFF_OCCA + (size_t)5 * HW * 4;
static constexpr size_t OFF_OCCC = OFF_OCCB + (size_t)3 * HW * 4;
static constexpr size_t OFF_AB   = OFF_OCCC + (size_t)2 * HW * 4;

extern "C" void kernel_launch(void* const* d_in, const int* in_sizes, int n_in,
                              void* d_out, int out_size, void* d_ws, size_t ws_size,
                              hipStream_t stream)
{
    (void)in_sizes; (void)n_in; (void)out_size; (void)ws_size;

    const float*         x    = (const float*)d_in[0];
    const unsigned char* mask = (const unsigned char*)d_in[1];
    const float* w[3]  = { (const float*)d_in[2],  (const float*)d_in[8],  (const float*)d_in[14] };
    const float* b[3]  = { (const float*)d_in[3],  (const float*)d_in[9],  (const float*)d_in[15] };
    const float* g[3]  = { (const float*)d_in[4],  (const float*)d_in[10], (const float*)d_in[16] };
    const float* be[3] = { (const float*)d_in[5],  (const float*)d_in[11], (const float*)d_in[17] };
    const float* rm[3] = { (const float*)d_in[6],  (const float*)d_in[12], (const float*)d_in[18] };
    const float* rv[3] = { (const float*)d_in[7],  (const float*)d_in[13], (const float*)d_in[19] };

    char* ws = (char*)d_ws;
    __bf16* XP   = (__bf16*)(ws + OFF_XP);
    __bf16* Y0P  = (__bf16*)(ws + OFF_Y0P);
    __bf16* Y1P  = (__bf16*)(ws + OFF_Y1P);
    __bf16* WT[3]= { (__bf16*)(ws + OFF_W0T), (__bf16*)(ws + OFF_W1T), (__bf16*)(ws + OFF_W2T) };
    float* OCC0  = (float*)(ws + OFF_OCC0);
    float* OCCA  = (float*)(ws + OFF_OCCA);
    float* OCCB  = (float*)(ws + OFF_OCCB);
    float* OCCC  = (float*)(ws + OFF_OCCC);
    float* ALPHA = (float*)(ws + OFF_AB);          // [3][64]
    float* BETA  = ALPHA + 3 * 64;                 // [3][64]

    const int cin[3] = { 128, 64, 64 };

    // --- small prep kernels ---
    for (int l = 0; l < 3; ++l) {
        prep_bn<<<1, 64, 0, stream>>>(b[l], g[l], be[l], rm[l], rv[l],
                                      ALPHA + 64 * l, BETA + 64 * l);
        int wtot = 27 * COUT * cin[l];
        prep_w<<<(wtot + 255) / 256, 256, 0, stream>>>(w[l], WT[l], cin[l]);
    }
    mask_to_occ<<<(10 * HW + 255) / 256, 256, 0, stream>>>(mask, OCC0, 10 * HW);
    occ_prop<<<(5 * HW + 255) / 256, 256, 0, stream>>>(OCC0, OCCA, 10, 5, 2, 1);
    occ_prop<<<(3 * HW + 255) / 256, 256, 0, stream>>>(OCCA, OCCB, 5, 3, 1, 0);
    occ_prop<<<(2 * HW + 255) / 256, 256, 0, stream>>>(OCCB, OCCC, 3, 2, 2, 1);
    prep_x_pad<<<(10 * HWP + 255) / 256, 256, 0, stream>>>(x, mask, XP, 10);

    // zero the padded intermediates once so their halos are exact zeros
    {
        int n16 = (int)(SZ_Y0P / 16);
        zero_b128<<<(n16 + 255) / 256, 256, 0, stream>>>((v4u*)Y0P, n16);
        n16 = (int)(SZ_Y1P / 16);
        zero_b128<<<(n16 + 255) / 256, 256, 0, stream>>>((v4u*)Y1P, n16);
    }

    // --- the three WMMA conv layers (fully compile-time geometry) ---
    dim3 blk(256);
    conv3x3x3_wmma<4, 10, 5, 2, 1, false><<<dim3(6, H, 5), blk, 0, stream>>>(
        XP, WT[0], ALPHA + 0,   BETA + 0,   OCCA, Y0P);
    conv3x3x3_wmma<2, 5, 3, 1, 0, false><<<dim3(6, H, 3), blk, 0, stream>>>(
        Y0P, WT[1], ALPHA + 64, BETA + 64,  OCCB, Y1P);
    conv3x3x3_wmma<2, 3, 2, 2, 1, true><<<dim3(6, H, 2), blk, 0, stream>>>(
        Y1P, WT[2], ALPHA + 128, BETA + 128, OCCC, d_out);
}
